// MultiHeadedAttention_73400991089329
// MI455X (gfx1250) — matrix operand order repaired and checked by reading, not compiled, not run on hardware
//
#include <hip/hip_runtime.h>

// -------- CDNA5 (gfx1250) multi-head attention, bf16 WMMA pipeline --------
// B=2, S=2048, D_MODEL=1024, H=16, HEAD=64.

typedef __bf16 bf16;
typedef __attribute__((ext_vector_type(16))) __bf16 v16bf;
typedef __attribute__((ext_vector_type(8)))  float  v8f;

#define MHA_B 2
#define MHA_S 2048
#define MHA_D 1024
#define MHA_H 16
#define MHA_HD 64
#define MHA_M (MHA_B * MHA_S)   // 4096 rows for the projections

// gfx1250 async global->LDS staging if the toolchain exposes it
#if defined(__has_builtin)
#if __has_builtin(__builtin_amdgcn_global_load_async_to_lds_b128) && \
    __has_builtin(__builtin_amdgcn_s_wait_asynccnt)
#define MHA_ASYNC_LDS 1
#endif
#endif
#ifndef MHA_ASYNC_LDS
#define MHA_ASYNC_LDS 0
#endif

#if MHA_ASYNC_LDS
typedef int mha_v4i __attribute__((ext_vector_type(4)));
typedef __attribute__((address_space(1))) mha_v4i mha_g_v4i;   // global (prints as __device__)
typedef __attribute__((address_space(3))) mha_v4i mha_l_v4i;   // LDS (prints as __shared__)

// async 16-byte global -> LDS copy (per-lane), tracked by ASYNCcnt
__device__ __forceinline__ void mha_async_cp16(const bf16* gsrc, bf16* lds_dst) {
  __builtin_amdgcn_global_load_async_to_lds_b128(
      (mha_g_v4i*)gsrc, (mha_l_v4i*)lds_dst, 0, 0);
}
#endif

// ---- helpers -------------------------------------------------------------

__device__ __forceinline__ bf16 f2bf(float f) {
  union { float f; unsigned u; } v; v.f = f;
  unsigned r = v.u + 0x7fffu + ((v.u >> 16) & 1u);   // round-to-nearest-even
  unsigned short hs = (unsigned short)(r >> 16);
  bf16 out;
  __builtin_memcpy(&out, &hs, sizeof(out));
  return out;
}

__device__ __forceinline__ v8f wmma_bf16(v16bf a, v16bf b, v8f c) {
  // D(f32 16x16) = A(bf16 16x32) * B(bf16 32x16) + C
  return __builtin_amdgcn_wmma_f32_16x16x32_bf16(
      /*neg_a=*/false, a, /*neg_b=*/false, b,
      /*c_mod=*/(short)0, c, /*reuse_a=*/false, /*reuse_b=*/false);
}

// A fragment (16x32 bf16) from row-major memory/LDS with leading dim `lda`.
// Lanes 0-15: row=lane, K {0..7,16..23}; lanes 16-31: row=lane-16, K {8..15,24..31}.
// Per-lane accesses are two contiguous 16-byte runs -> b128 loads.
__device__ __forceinline__ v16bf load_a_rm(const bf16* base, int lda, int lane) {
  const bool hi = lane >= 16;
  const int row = hi ? lane - 16 : lane;
  const int kb  = hi ? 8 : 0;
  const bf16* p = base + row * lda + kb;
  v16bf a;
#pragma unroll
  for (int i = 0; i < 8; ++i) a[i] = p[i];
#pragma unroll
  for (int i = 0; i < 8; ++i) a[8 + i] = p[16 + i];
  return a;
}

// B fragment (32x16 bf16) from K-contiguous storage: element (k,n) at base[n*ldk + k].
// Lanes 0-15: col=lane, K {0..7,16..23}; lanes 16-31: col=lane-16, K {8..15,24..31}.
// Per-lane accesses are two contiguous 16-byte runs -> b128 loads.
__device__ __forceinline__ v16bf load_b_kc(const bf16* base, size_t ldk, int lane) {
  const int col = lane & 15;
  const int kb  = (lane >= 16) ? 8 : 0;
  const bf16* p = base + (size_t)col * ldk + kb;
  v16bf bm;
#pragma unroll
  for (int i = 0; i < 8; ++i) bm[i] = p[i];
#pragma unroll
  for (int i = 0; i < 8; ++i) bm[8 + i] = p[16 + i];
  return bm;
}

// ---- kernel 0a: plain f32 -> bf16 conversion -----------------------------

__global__ void __launch_bounds__(256) mha_cvt_bf16(const float* __restrict__ src,
                                                    bf16* __restrict__ dst, int n) {
  int i = blockIdx.x * 256 + threadIdx.x;
  if (i < n) dst[i] = f2bf(src[i]);
}

// ---- kernel 0b: f32 [K][N] -> bf16 [N][K] transposed conversion ----------
// 32x32 tiles through LDS; grid (32,32), block 256.

__global__ void __launch_bounds__(256) mha_cvt_bf16_T(const float* __restrict__ src,
                                                      bf16* __restrict__ dst) {
  __shared__ float tile[32][33];
  const int c0 = blockIdx.x * 32;
  const int r0 = blockIdx.y * 32;
  const int tx = threadIdx.x & 31;
  const int ty = threadIdx.x >> 5;   // 0..7
#pragma unroll
  for (int i = 0; i < 32; i += 8)
    tile[ty + i][tx] = src[(size_t)(r0 + ty + i) * MHA_D + c0 + tx];
  __syncthreads();
#pragma unroll
  for (int i = 0; i < 32; i += 8)
    dst[(size_t)(c0 + ty + i) * MHA_D + r0 + tx] = f2bf(tile[tx][ty + i]);
}

// ---- kernel 1: fused QKV projection (GEMM + bias, Q pre-scaled) ----------
// Block: 256 thr (8 waves). Block tile 128(M) x 64(N). Wave tile 32x32.
// Weights are pre-transposed [N][K] so both staging and B-fragments are contiguous.
// grid = (N/64=16, M/128=32, 3 projections)

__global__ void __launch_bounds__(256) mha_qkv_gemm(
    const bf16* __restrict__ Xb,
    const bf16* __restrict__ Wqt, const bf16* __restrict__ Wkt, const bf16* __restrict__ Wvt,
    const float* __restrict__ bq, const float* __restrict__ bk, const float* __restrict__ bv,
    bf16* __restrict__ Qb, bf16* __restrict__ Kb, bf16* __restrict__ Vb) {
  __shared__ bf16 Xs[128 * 32];   // [m][k] row-major
  __shared__ bf16 Ws[64 * 32];    // [n][k] (K-contiguous for B-fragments)

  const int z = blockIdx.z;
  const bf16* Wt = (z == 0) ? Wqt : (z == 1) ? Wkt : Wvt;
  const float* bias = (z == 0) ? bq : (z == 1) ? bk : bv;
  bf16* outp = (z == 0) ? Qb : (z == 1) ? Kb : Vb;
  const float scale = (z == 0) ? 0.125f : 1.0f;   // 1/sqrt(64) folded into Q

  const int n0 = blockIdx.x * 64;
  const int m0 = blockIdx.y * 128;
  const int t = threadIdx.x;
  const int lane = t & 31;
  const int wave = t >> 5;           // 0..7
  const int wm = (wave >> 1) * 32;   // wave M offset (0,32,64,96)
  const int wn = (wave & 1) * 32;    // wave N offset (0,32)

  v8f c00 = {}, c01 = {}, c10 = {}, c11 = {};

  for (int k0 = 0; k0 < MHA_D; k0 += 32) {
#if MHA_ASYNC_LDS
    // X tile: 8 KiB = 512 x 16B chunks (2/thread); W tile: 4 KiB (1/thread)
#pragma unroll
    for (int e = 0; e < 2; ++e) {
      int eo = (t * 2 + e) * 8;            // element offset in Xs
      int row = eo >> 5, col = eo & 31;
      mha_async_cp16(&Xb[(size_t)(m0 + row) * MHA_D + k0 + col], &Xs[eo]);
    }
    {
      int eo = t * 8;
      int row = eo >> 5, kk = eo & 31;
      mha_async_cp16(&Wt[(size_t)(n0 + row) * MHA_D + k0 + kk], &Ws[eo]);
    }
    __builtin_amdgcn_s_wait_asynccnt(0);
#else
#pragma unroll
    for (int e = 0; e < 16; ++e) {
      int idx = t * 16 + e;                  // 0..4095
      int row = idx >> 5, col = idx & 31;
      Xs[idx] = Xb[(size_t)(m0 + row) * MHA_D + k0 + col];
    }
#pragma unroll
    for (int e = 0; e < 8; ++e) {
      int idx = t * 8 + e;                   // 0..2047
      int row = idx >> 5, kk = idx & 31;
      Ws[idx] = Wt[(size_t)(n0 + row) * MHA_D + k0 + kk];
    }
#endif
    __syncthreads();

    v16bf a0 = load_a_rm(&Xs[(wm + 0) * 32], 32, lane);
    v16bf a1 = load_a_rm(&Xs[(wm + 16) * 32], 32, lane);
    v16bf b0 = load_b_kc(&Ws[(wn + 0) * 32], 32, lane);
    v16bf b1 = load_b_kc(&Ws[(wn + 16) * 32], 32, lane);
    c00 = wmma_bf16(a0, b0, c00);
    c01 = wmma_bf16(a0, b1, c01);
    c10 = wmma_bf16(a1, b0, c10);
    c11 = wmma_bf16(a1, b1, c11);
    __syncthreads();
  }

  // epilogue: bias + scale. Q,K -> [B,H,S,64]; V -> [B,H,64,S] (head-transposed)
  const int N = lane & 15;
  const int Mb = (lane >= 16) ? 8 : 0;
#pragma unroll
  for (int fm = 0; fm < 2; ++fm) {
#pragma unroll
    for (int fn = 0; fn < 2; ++fn) {
      const v8f& c = (fm == 0) ? ((fn == 0) ? c00 : c01) : ((fn == 0) ? c10 : c11);
#pragma unroll
      for (int r = 0; r < 8; ++r) {
        int m = m0 + wm + fm * 16 + Mb + r;
        int n = n0 + wn + fn * 16 + N;
        float v = (c[r] + bias[n]) * scale;
        int bb = m >> 11, s = m & 2047, h = n >> 6, d = n & 63;
        if (z == 2) {
          // contiguous in s per lane across r
          outp[(((size_t)(bb * MHA_H + h) * MHA_HD + d) * MHA_S) + s] = f2bf(v);
        } else {
          outp[(((size_t)(bb * MHA_H + h) * MHA_S + s) * MHA_HD) + d] = f2bf(v);
        }
      }
    }
  }
}

// ---- kernel 2: flash attention, 1 wave per (b, h, 16 query rows) ---------
// 64 iterations x 32 keys: 4 WMMA for Q*K^T + online softmax + 4 WMMA for P*V.
// K stored [S,64] makes K^T B-fragments contiguous; V stored [64,S] makes
// V B-fragments contiguous -> all global traffic is b128-class.

__global__ void __launch_bounds__(32) mha_flash(
    const bf16* __restrict__ Q, const bf16* __restrict__ K,
    const bf16* __restrict__ V, bf16* __restrict__ Ctx) {
  __shared__ bf16 Pl[16 * 32];   // P tile staged row-major [row][k]

  const int lane = threadIdx.x;
  const int qb = blockIdx.x;     // 0..127 (16 query rows each)
  const int h  = blockIdx.y;     // 0..15
  const int b  = blockIdx.z;     // 0..1

  const size_t head_base = (size_t)(b * MHA_H + h) * MHA_S * MHA_HD;
  const bf16* Qp = Q + head_base + (size_t)qb * 16 * MHA_HD;
  const bf16* Kp = K + head_base;
  const bf16* Vp = V + head_base;   // [64][2048]

  // Q fragments held in registers for the whole pass (K-dim 64 = 2 chunks)
  v16bf aQ0 = load_a_rm(Qp, MHA_HD, lane);
  v16bf aQ1 = load_a_rm(Qp + 32, MHA_HD, lane);

  v8f o0 = {}, o1 = {}, o2 = {}, o3 = {};
  float mrow[8], lrow[8];
#pragma unroll
  for (int r = 0; r < 8; ++r) { mrow[r] = -3.0e38f; lrow[r] = 0.0f; }

  for (int j = 0; j < 64; ++j) {
    const int key0 = j * 32;
    const bf16* Kt = Kp + (size_t)key0 * MHA_HD;
    if (j < 63) {   // pull next tiles toward the WGP (global_prefetch_b8)
      __builtin_prefetch(Kt + 32 * MHA_HD, 0, 1);
      __builtin_prefetch(Vp + (size_t)lane * MHA_S + key0 + 32, 0, 1);
      __builtin_prefetch(Vp + (size_t)(lane + 32) * MHA_S + key0 + 32, 0, 1);
    }

    // scores tile 16x32 (Q pre-scaled by 1/sqrt(d)); K^T frags contiguous (ldk=64)
    v8f s0 = {}, s1 = {};
    s0 = wmma_bf16(aQ0, load_b_kc(Kt + 0,            MHA_HD, lane), s0);
    s0 = wmma_bf16(aQ1, load_b_kc(Kt + 32,           MHA_HD, lane), s0);
    s1 = wmma_bf16(aQ0, load_b_kc(Kt + 16 * MHA_HD,      MHA_HD, lane), s1);
    s1 = wmma_bf16(aQ1, load_b_kc(Kt + 16 * MHA_HD + 32, MHA_HD, lane), s1);

    // online softmax: each row lives in one VGPR across a 16-lane half
    float fr[8];
#pragma unroll
    for (int r = 0; r < 8; ++r) {
      float t = fmaxf(s0[r], s1[r]);
      t = fmaxf(t, __shfl_xor(t, 1, 32));
      t = fmaxf(t, __shfl_xor(t, 2, 32));
      t = fmaxf(t, __shfl_xor(t, 4, 32));
      t = fmaxf(t, __shfl_xor(t, 8, 32));
      float mn = fmaxf(mrow[r], t);
      fr[r] = __expf(mrow[r] - mn);
      mrow[r] = mn;
      float p0 = __expf(s0[r] - mn);
      float p1 = __expf(s1[r] - mn);
      s0[r] = p0; s1[r] = p1;
      float sum = p0 + p1;
      sum += __shfl_xor(sum, 1, 32);
      sum += __shfl_xor(sum, 2, 32);
      sum += __shfl_xor(sum, 4, 32);
      sum += __shfl_xor(sum, 8, 32);
      lrow[r] = lrow[r] * fr[r] + sum;
      o0[r] *= fr[r]; o1[r] *= fr[r]; o2[r] *= fr[r]; o3[r] *= fr[r];
    }

    // stage P (bf16) row-major through LDS; reload as WMMA A via b128s
    {
      const int col = lane & 15;
      const int rb  = (lane >= 16) ? 8 : 0;
#pragma unroll
      for (int r = 0; r < 8; ++r) {
        Pl[(rb + r) * 32 + col]      = f2bf(s0[r]);
        Pl[(rb + r) * 32 + 16 + col] = f2bf(s1[r]);
      }
    }
    __syncthreads();
    v16bf aP = load_a_rm(Pl, 32, lane);

    // O += P(16x32) * V(32x64); V frags contiguous (ldk=2048)
    o0 = wmma_bf16(aP, load_b_kc(Vp + (size_t)0  * MHA_S + key0, MHA_S, lane), o0);
    o1 = wmma_bf16(aP, load_b_kc(Vp + (size_t)16 * MHA_S + key0, MHA_S, lane), o1);
    o2 = wmma_bf16(aP, load_b_kc(Vp + (size_t)32 * MHA_S + key0, MHA_S, lane), o2);
    o3 = wmma_bf16(aP, load_b_kc(Vp + (size_t)48 * MHA_S + key0, MHA_S, lane), o3);
    __syncthreads();
  }

  // finalize: divide by row sums, merge heads into [B,S,D] bf16
  const int N = lane & 15;
  const int Mb = (lane >= 16) ? 8 : 0;
  const int row0 = qb * 16;
#pragma unroll
  for (int r = 0; r < 8; ++r) {
    float rinv = 1.0f / lrow[r];
    size_t rowaddr = ((size_t)(b * MHA_S + row0 + Mb + r)) * MHA_D + h * MHA_HD;
    Ctx[rowaddr + 0  + N] = f2bf(o0[r] * rinv);
    Ctx[rowaddr + 16 + N] = f2bf(o1[r] * rinv);
    Ctx[rowaddr + 32 + N] = f2bf(o2[r] * rinv);
    Ctx[rowaddr + 48 + N] = f2bf(o3[r] * rinv);
  }
}

// ---- kernel 3: output projection (GEMM + bias -> f32 d_out) --------------

__global__ void __launch_bounds__(256) mha_out_gemm(
    const bf16* __restrict__ Ctx, const bf16* __restrict__ Wot,
    const float* __restrict__ bo, float* __restrict__ out) {
  __shared__ bf16 Xs[128 * 32];
  __shared__ bf16 Ws[64 * 32];   // [n][k]

  const int n0 = blockIdx.x * 64;
  const int m0 = blockIdx.y * 128;
  const int t = threadIdx.x;
  const int lane = t & 31;
  const int wave = t >> 5;
  const int wm = (wave >> 1) * 32;
  const int wn = (wave & 1) * 32;

  v8f c00 = {}, c01 = {}, c10 = {}, c11 = {};

  for (int k0 = 0; k0 < MHA_D; k0 += 32) {
#if MHA_ASYNC_LDS
#pragma unroll
    for (int e = 0; e < 2; ++e) {
      int eo = (t * 2 + e) * 8;
      int row = eo >> 5, col = eo & 31;
      mha_async_cp16(&Ctx[(size_t)(m0 + row) * MHA_D + k0 + col], &Xs[eo]);
    }
    {
      int eo = t * 8;
      int row = eo >> 5, kk = eo & 31;
      mha_async_cp16(&Wot[(size_t)(n0 + row) * MHA_D + k0 + kk], &Ws[eo]);
    }
    __builtin_amdgcn_s_wait_asynccnt(0);
#else
#pragma unroll
    for (int e = 0; e < 16; ++e) {
      int idx = t * 16 + e;
      int row = idx >> 5, col = idx & 31;
      Xs[idx] = Ctx[(size_t)(m0 + row) * MHA_D + k0 + col];
    }
#pragma unroll
    for (int e = 0; e < 8; ++e) {
      int idx = t * 8 + e;
      int row = idx >> 5, kk = idx & 31;
      Ws[idx] = Wot[(size_t)(n0 + row) * MHA_D + k0 + kk];
    }
#endif
    __syncthreads();

    v16bf a0 = load_a_rm(&Xs[(wm + 0) * 32], 32, lane);
    v16bf a1 = load_a_rm(&Xs[(wm + 16) * 32], 32, lane);
    v16bf b0 = load_b_kc(&Ws[(wn + 0) * 32], 32, lane);
    v16bf b1 = load_b_kc(&Ws[(wn + 16) * 32], 32, lane);
    c00 = wmma_bf16(a0, b0, c00);
    c01 = wmma_bf16(a0, b1, c01);
    c10 = wmma_bf16(a1, b0, c10);
    c11 = wmma_bf16(a1, b1, c11);
    __syncthreads();
  }

  const int N = lane & 15;
  const int Mb = (lane >= 16) ? 8 : 0;
#pragma unroll
  for (int fm = 0; fm < 2; ++fm) {
#pragma unroll
    for (int fn = 0; fn < 2; ++fn) {
      const v8f& c = (fm == 0) ? ((fn == 0) ? c00 : c01) : ((fn == 0) ? c10 : c11);
#pragma unroll
      for (int r = 0; r < 8; ++r) {
        int m = m0 + wm + fm * 16 + Mb + r;
        int n = n0 + wn + fn * 16 + N;
        out[(size_t)m * MHA_D + n] = c[r] + bo[n];
      }
    }
  }
}

// ---- launch --------------------------------------------------------------

extern "C" void kernel_launch(void* const* d_in, const int* in_sizes, int n_in,
                              void* d_out, int out_size, void* d_ws, size_t ws_size,
                              hipStream_t stream) {
  const float* X  = (const float*)d_in[0];
  const float* Wq = (const float*)d_in[1];
  const float* bq = (const float*)d_in[2];
  const float* Wk = (const float*)d_in[3];
  const float* bk = (const float*)d_in[4];
  const float* Wv = (const float*)d_in[5];
  const float* bv = (const float*)d_in[6];
  const float* Wo = (const float*)d_in[7];
  const float* bo = (const float*)d_in[8];
  float* out = (float*)d_out;

  // workspace layout (bf16 elements): 48 MiB total
  bf16* Xb  = (bf16*)d_ws;                         // 4096x1024
  bf16* Wqt = Xb  + (size_t)MHA_M * MHA_D;         // transposed weights, 1024x1024 each
  bf16* Wkt = Wqt + (size_t)MHA_D * MHA_D;
  bf16* Wvt = Wkt + (size_t)MHA_D * MHA_D;
  bf16* Wot = Wvt + (size_t)MHA_D * MHA_D;
  bf16* Qb  = Wot + (size_t)MHA_D * MHA_D;         // [B,H,S,64]
  bf16* Kb  = Qb  + (size_t)MHA_M * MHA_D;         // [B,H,S,64]
  bf16* Vb  = Kb  + (size_t)MHA_M * MHA_D;         // [B,H,64,S]
  bf16* Ctx = Vb  + (size_t)MHA_M * MHA_D;         // [B,S,D]

  const int nX = MHA_M * MHA_D;      // 4194304
  mha_cvt_bf16<<<(nX + 255) / 256, 256, 0, stream>>>(X, Xb, nX);
  mha_cvt_bf16_T<<<dim3(32, 32), 256, 0, stream>>>(Wq, Wqt);
  mha_cvt_bf16_T<<<dim3(32, 32), 256, 0, stream>>>(Wk, Wkt);
  mha_cvt_bf16_T<<<dim3(32, 32), 256, 0, stream>>>(Wv, Wvt);
  mha_cvt_bf16_T<<<dim3(32, 32), 256, 0, stream>>>(Wo, Wot);

  mha_qkv_gemm<<<dim3(MHA_D / 64, MHA_M / 128, 3), 256, 0, stream>>>(
      Xb, Wqt, Wkt, Wvt, bq, bk, bv, Qb, Kb, Vb);

  mha_flash<<<dim3(MHA_S / 16, MHA_H, MHA_B), 32, 0, stream>>>(Qb, Kb, Vb, Ctx);

  mha_out_gemm<<<dim3(MHA_D / 64, MHA_M / 128), 256, 0, stream>>>(Ctx, Wot, bo, out);
}
